// MultiHeadSelfAttention_25933012533694
// MI455X (gfx1250) — matrix operand order repaired and checked by reading, not compiled, not run on hardware
//
#include <hip/hip_runtime.h>
#include <hip/hip_bf16.h>

// ---------------------------------------------------------------------------
// MI455X (gfx1250) multi-head self-attention, bf16 WMMA + fp32 exp2 softmax.
//   B=4, S=2048, D_MODEL=1024, H=16, Dk=64
// ---------------------------------------------------------------------------

typedef __bf16 bf16;
typedef __attribute__((ext_vector_type(16))) __bf16 bf16x16;
typedef __attribute__((ext_vector_type(8)))  __bf16 bf16x8;
typedef __attribute__((ext_vector_type(4)))  __bf16 bf16x4;
typedef __attribute__((ext_vector_type(8)))  float  f32x8;
typedef __attribute__((ext_vector_type(4)))  float  f32x4;

#define DMODEL 1024
#define SEQ    2048
#define NH     16
#define DK     64
#define BATCH  4
#define MWORDS (SEQ / 32)                  // mask bit-words per row

// softmax scale folded into Q projection, in log2 domain:
// exp(qk/8) == exp2(qk * 0.125 * log2(e))
#define SCALE_Q 0.18033688011112042f
#define NEGINF  -1.0e30f

union Frag16 { bf16x16 v; bf16x8 h[2]; };

__device__ __forceinline__ f32x8 zero8() {
    f32x8 z = {0.f,0.f,0.f,0.f,0.f,0.f,0.f,0.f};
    return z;
}

// native v_exp_f32 (no libm range fixup; args are <= 0 here and extreme
// inputs only occur where the weight vanishes anyway)
__device__ __forceinline__ float fast_exp2(float x) {
#if __has_builtin(__builtin_amdgcn_exp2f)
    return __builtin_amdgcn_exp2f(x);
#else
    return exp2f(x);
#endif
}

__device__ __forceinline__ float fast_rcp(float x) {
#if __has_builtin(__builtin_amdgcn_rcpf)
    return __builtin_amdgcn_rcpf(x);
#else
    return 1.0f / x;
#endif
}

// ---------------------------------------------------------------------------
// f32 -> bf16 conversion (vectorized x4)
// ---------------------------------------------------------------------------
__global__ __launch_bounds__(256) void cvt_f32_bf16(const float* __restrict__ in,
                                                    bf16* __restrict__ out, int n4) {
    int i = blockIdx.x * blockDim.x + threadIdx.x;
    if (i < n4) {
        f32x4 v = ((const f32x4*)in)[i];
        bf16x4 o;
        o.x = (bf16)v.x; o.y = (bf16)v.y; o.z = (bf16)v.z; o.w = (bf16)v.w;
        ((bf16x4*)out)[i] = o;
    }
}

// ---------------------------------------------------------------------------
// Pack int mask [S,S] into bitmask (bit k of word = mask element != 0).
// wave32 __ballot: lane l <-> bit l.
// ---------------------------------------------------------------------------
__global__ __launch_bounds__(256) void pack_mask(const int* __restrict__ mask,
                                                 unsigned* __restrict__ mbits) {
    const int i = blockIdx.x * blockDim.x + threadIdx.x;
    const unsigned long long b = __ballot(mask[i] != 0);
    if ((threadIdx.x & 31) == 0) mbits[i >> 5] = (unsigned)b;
}

// ---------------------------------------------------------------------------
// Per-row "fully open" flag: rowfull[q] = 1 iff all 2048 mask bits of row q
// are set. One wave per row (64 words, 2 per lane).
// ---------------------------------------------------------------------------
__global__ __launch_bounds__(256) void mask_rowfull(const unsigned* __restrict__ mbits,
                                                    int* __restrict__ rowfull) {
    const int lane = threadIdx.x & 31;
    const int q = (blockIdx.x * blockDim.x + threadIdx.x) >> 5;
    const unsigned* row = mbits + (size_t)q * MWORDS;
    const int ok = (row[lane] == 0xFFFFFFFFu) && (row[lane + 32] == 0xFFFFFFFFu);
    const int all = __all(ok);
    if (lane == 0) rowfull[q] = all;
}

// ---------------------------------------------------------------------------
// bf16 NT GEMM: C[M,N] = A[M,K] * Bw[N,K]^T + bias, K = N = 1024, M = 8192.
// 256 thr / 8 waves, block tile 128x128, wave tile 32x64, k-step 64,
// double-buffered LDS (one barrier per k-step).
// MODE 0: Q  -> bf16 [B,H,S,Dk], scaled by SCALE_Q
// MODE 1: K  -> bf16 [B,H,S,Dk]
// MODE 2: V  -> bf16 [B,H,Dk,S] (pre-transposed for attention B-fragments)
// MODE 3: out-> f32  [M,N]
// ---------------------------------------------------------------------------
template<int MODE>
__global__ __launch_bounds__(256) void gemm_bf16(const bf16* __restrict__ A,
                                                 const bf16* __restrict__ Bw,
                                                 const float* __restrict__ bias,
                                                 void* __restrict__ Cout) {
    constexpr int K = 1024, N = 1024, KS = 64;
    constexpr int LDT = 72;                 // padded LDS row stride (bf16)
    __shared__ bf16 At[2][128 * LDT];
    __shared__ bf16 Bt[2][128 * LDT];

    const int n0   = blockIdx.x * 128;
    const int m0   = blockIdx.y * 128;
    const int tid  = threadIdx.x;
    const int lane = tid & 31;
    const int w    = tid >> 5;
    const int wm   = w >> 1;                // 0..3
    const int wn   = w & 1;                 // 0..1
    const int lr   = lane & 15;
    const int lh   = lane >> 4;

    f32x8 acc[2][4];
#pragma unroll
    for (int i = 0; i < 2; i++)
#pragma unroll
        for (int j = 0; j < 4; j++) acc[i][j] = zero8();

    // cooperative staging: thread owns 32 bf16 of one 64-wide k-slab row
    const int srow = tid >> 1;              // 0..127
    const int sc0  = (tid & 1) * 32;        // 0 or 32
    const bf16* gA = A  + (size_t)(m0 + srow) * K + sc0;
    const bf16* gB = Bw + (size_t)(n0 + srow) * K + sc0;
    const int soff = srow * LDT + sc0;

    bf16x8 ra[4], rb[4];
    auto loadSlab = [&](int k0) {
#pragma unroll
        for (int c = 0; c < 4; c++) {
            ra[c] = *(const bf16x8*)(gA + k0 + c * 8);
            rb[c] = *(const bf16x8*)(gB + k0 + c * 8);
        }
    };
    auto storeSlab = [&](int buf) {
#pragma unroll
        for (int c = 0; c < 4; c++) {
            *(bf16x8*)(At[buf] + soff + c * 8) = ra[c];
            *(bf16x8*)(Bt[buf] + soff + c * 8) = rb[c];
        }
    };

    loadSlab(0);
    storeSlab(0);

    for (int i = 0; i < K / KS; i++) {
        __syncthreads();                    // buf[i&1] ready for all waves
        if (i + 1 < K / KS) loadSlab((i + 1) * KS);   // overlap with WMMA

        const bf16* ab = At[i & 1];
        const bf16* bb = Bt[i & 1];
#pragma unroll
        for (int kk = 0; kk < 2; kk++) {
            Frag16 af[2];
#pragma unroll
            for (int ms = 0; ms < 2; ms++) {
                const bf16* p = ab + (wm * 32 + ms * 16 + lr) * LDT + kk * 32 + lh * 8;
                af[ms].h[0] = *(const bf16x8*)p;
                af[ms].h[1] = *(const bf16x8*)(p + 16);
            }
#pragma unroll
            for (int ns = 0; ns < 4; ns++) {
                Frag16 bfr;
                const bf16* p = bb + (wn * 64 + ns * 16 + lr) * LDT + kk * 32 + lh * 8;
                bfr.h[0] = *(const bf16x8*)p;
                bfr.h[1] = *(const bf16x8*)(p + 16);
#pragma unroll
                for (int ms = 0; ms < 2; ms++)
                    acc[ms][ns] = __builtin_amdgcn_wmma_f32_16x16x32_bf16(
                        false, af[ms].v, false, bfr.v, (short)0, acc[ms][ns], false, false);
            }
        }
        if (i + 1 < K / KS) storeSlab((i + 1) & 1);
    }

    // epilogue
#pragma unroll
    for (int ms = 0; ms < 2; ms++) {
#pragma unroll
        for (int ns = 0; ns < 4; ns++) {
            const int gn = n0 + wn * 64 + ns * 16 + lr;
            const float bv = bias[gn];
#pragma unroll
            for (int r = 0; r < 8; r++) {
                const int gm = m0 + wm * 32 + ms * 16 + r + lh * 8;
                float val = acc[ms][ns][r] + bv;
                if (MODE == 0) val *= SCALE_Q;
                if (MODE == 3) {
                    ((float*)Cout)[(size_t)gm * N + gn] = val;
                } else {
                    const int b = gm >> 11, s = gm & (SEQ - 1);
                    const int h = gn >> 6,  d = gn & (DK - 1);
                    bf16* o = (bf16*)Cout;
                    if (MODE == 2)   // V: [B,H,Dk,S]
                        o[((size_t)(b * NH + h) * DK + d) * SEQ + s] = (bf16)val;
                    else             // Q/K: [B,H,S,Dk]
                        o[((size_t)(b * NH + h) * SEQ + s) * DK + d] = (bf16)val;
                }
            }
        }
    }
}

// ---------------------------------------------------------------------------
// Flash attention: grid (S/128, B*H), 256 thr, no block barriers in the loop.
// Wave owns 16 query rows; 64 keys/step: 8 WMMA (QK^T) + 8 WMMA (P*V).
// Wave-uniform loop unswitch: if all of the wave's mask rows are fully open,
// the hot loop carries no mask loads/selects at all. Running-stats update is
// unconditional (a per-step skip branch costs more in accumulator phi copies
// than the corr math it saves).
// ---------------------------------------------------------------------------
__global__ __launch_bounds__(256) void flash_attn(const bf16* __restrict__ Qb,
                                                  const bf16* __restrict__ Kb,
                                                  const bf16* __restrict__ Vt,
                                                  const unsigned* __restrict__ mbits,
                                                  const int* __restrict__ rowfull,
                                                  bf16* __restrict__ Ob) {
    constexpr int LDP = 72;                 // padded P-buffer row stride
    __shared__ bf16 pbuf[8][16 * LDP];

    const int tid  = threadIdx.x;
    const int lane = tid & 31;
    const int w    = tid >> 5;
    const int lr   = lane & 15;
    const int lh   = lane >> 4;
    const int bh   = blockIdx.y;
    const int b    = bh >> 4, hh = bh & 15;
    const int q0   = blockIdx.x * 128 + w * 16;

    const bf16* Q  = Qb + ((size_t)bh * SEQ + q0) * DK;
    const bf16* Kp = Kb + (size_t)bh * SEQ * DK;
    const bf16* Vp = Vt + (size_t)bh * DK * SEQ;
    const unsigned* mb = mbits + (size_t)(q0 + 8 * lh) * MWORDS;

    // wave-uniform: are all 16 of this wave's mask rows fully open?
    int fa = 1;
#pragma unroll
    for (int r = 0; r < 8; r++) fa &= rowfull[q0 + r + 8 * lh];
    const bool FULL = __all(fa);

    // persistent Q fragments
    Frag16 qf[2];
#pragma unroll
    for (int ks = 0; ks < 2; ks++) {
        const bf16* p = Q + lr * DK + ks * 32 + lh * 8;
        qf[ks].h[0] = *(const bf16x8*)p;
        qf[ks].h[1] = *(const bf16x8*)(p + 16);
    }

    f32x8 oacc[4];
#pragma unroll
    for (int i = 0; i < 4; i++) oacc[i] = zero8();
    float mrun[8], lrun[8];
#pragma unroll
    for (int r = 0; r < 8; r++) { mrun[r] = NEGINF; lrun[r] = 0.f; }

    bf16* myp = pbuf[w];

    for (int kb = 0; kb < SEQ; kb += 64) {
        // prefetch next K/V tiles (one wave per block; whole cachelines)
        if (w == 0 && kb + 64 < SEQ) {
            __builtin_prefetch(Kp + (size_t)(kb + 64 + lane) * DK, 0, 3);
            __builtin_prefetch(Kp + (size_t)(kb + 96 + lane) * DK, 0, 3);
            __builtin_prefetch(Vp + (size_t)lane * SEQ + kb + 64, 0, 3);
            __builtin_prefetch(Vp + (size_t)(lane + 32) * SEQ + kb + 64, 0, 3);
        }
        // ---- scores: four 16x16 key tiles, contraction d = 64 ----
        f32x8 sc[4];
#pragma unroll
        for (int t = 0; t < 4; t++) {
            sc[t] = zero8();
#pragma unroll
            for (int ks = 0; ks < 2; ks++) {
                Frag16 kf;
                const bf16* p = Kp + (size_t)(kb + t * 16 + lr) * DK + ks * 32 + lh * 8;
                kf.h[0] = *(const bf16x8*)p;
                kf.h[1] = *(const bf16x8*)(p + 16);
                sc[t] = __builtin_amdgcn_wmma_f32_16x16x32_bf16(
                    false, qf[ks].v, false, kf.v, (short)0, sc[t], false, false);
            }
        }
        // ---- masked online softmax over 64 keys (16-lane row groups) ----
        unsigned wA[8], wB[8];
        float corr[8];
        auto pass1 = [&](bool FULLP) {
#pragma unroll
            for (int r = 0; r < 8; r++) {
                float s0, s1, s2, s3;
                if (FULLP) {
                    s0 = sc[0][r]; s1 = sc[1][r]; s2 = sc[2][r]; s3 = sc[3][r];
                } else {
                    const unsigned t0 = wA[r] >> lr, t1 = wB[r] >> lr;
                    s0 = (t0 & 1u)         ? sc[0][r] : NEGINF;
                    s1 = ((t0 >> 16) & 1u) ? sc[1][r] : NEGINF;
                    s2 = (t1 & 1u)         ? sc[2][r] : NEGINF;
                    s3 = ((t1 >> 16) & 1u) ? sc[3][r] : NEGINF;
                }
                float mx = fmaxf(fmaxf(s0, s1), fmaxf(s2, s3));
#pragma unroll
                for (int d = 1; d < 16; d <<= 1) mx = fmaxf(mx, __shfl_xor(mx, d, 32));
                const float mn = fmaxf(mrun[r], mx);
                const float p0 = fast_exp2(s0 - mn);
                const float p1 = fast_exp2(s1 - mn);
                const float p2 = fast_exp2(s2 - mn);
                const float p3 = fast_exp2(s3 - mn);
                float rs = (p0 + p1) + (p2 + p3);
#pragma unroll
                for (int d = 1; d < 16; d <<= 1) rs += __shfl_xor(rs, d, 32);
                const float c = fast_exp2(mrun[r] - mn);
                corr[r] = c;
                lrun[r] = lrun[r] * c + rs;
                mrun[r] = mn;
                bf16* pr = myp + (r + lh * 8) * LDP + lr;
                pr[0]  = (bf16)p0;
                pr[16] = (bf16)p1;
                pr[32] = (bf16)p2;
                pr[48] = (bf16)p3;
            }
        };
        if (FULL) {
            pass1(true);
        } else {
            const int kw = kb >> 5;
#pragma unroll
            for (int r = 0; r < 8; r++) {
                wA[r] = mb[r * MWORDS + kw];
                wB[r] = mb[r * MWORDS + kw + 1];
            }
            pass1(false);
        }
        // ---- rescale running output (corr == 1 is the common, cheap case) --
#pragma unroll
        for (int nt = 0; nt < 4; nt++)
#pragma unroll
            for (int r = 0; r < 8; r++) oacc[nt][r] *= corr[r];
        // ---- P (16x64) as two A-fragments from wave-private LDS ----
        Frag16 pf[2];
#pragma unroll
        for (int kk = 0; kk < 2; kk++) {
            const bf16* p = myp + lr * LDP + kk * 32 + lh * 8;
            pf[kk].h[0] = *(const bf16x8*)p;
            pf[kk].h[1] = *(const bf16x8*)(p + 16);
        }
        // ---- O += P * V ----
#pragma unroll
        for (int nt = 0; nt < 4; nt++) {
#pragma unroll
            for (int kk = 0; kk < 2; kk++) {
                Frag16 vf;
                const bf16* p = Vp + (size_t)(nt * 16 + lr) * SEQ + kb + kk * 32 + lh * 8;
                vf.h[0] = *(const bf16x8*)p;
                vf.h[1] = *(const bf16x8*)(p + 16);
                oacc[nt] = __builtin_amdgcn_wmma_f32_16x16x32_bf16(
                    false, pf[kk].v, false, vf.v, (short)0, oacc[nt], false, false);
            }
        }
    }

    // ---- finalize: O /= l, write bf16 [B,S,H*Dk] ----
#pragma unroll
    for (int r = 0; r < 8; r++) {
        const float rl = fast_rcp(lrun[r]);
        const size_t base = ((size_t)b * SEQ + q0 + r + lh * 8) * DMODEL + hh * DK + lr;
#pragma unroll
        for (int nt = 0; nt < 4; nt++)
            Ob[base + nt * 16] = (bf16)(oacc[nt][r] * rl);
    }
}

// ---------------------------------------------------------------------------
// Workspace layout (bytes):
//   xb  @  0 MB : 16 MB   wqb @ 16 MB, wkb @ 18 MB, wvb @ 20 MB, wob @ 22 MB
//   Qb  @ 24 MB : 16 MB   Kb @ 40 MB : 16 MB   Vt @ 56 MB : 16 MB
//   aob @ 72 MB : 16 MB   mbits @ 88 MB : 0.5 MB   rowfull @ 88.5 MB : 8 KB
// ---------------------------------------------------------------------------
extern "C" void kernel_launch(void* const* d_in, const int* in_sizes, int n_in,
                              void* d_out, int out_size, void* d_ws, size_t ws_size,
                              hipStream_t stream) {
    const float* x    = (const float*)d_in[0];
    const int*   mask = (const int*)d_in[1];
    const float* wq   = (const float*)d_in[2];
    const float* bq   = (const float*)d_in[3];
    const float* wk   = (const float*)d_in[4];
    const float* bk   = (const float*)d_in[5];
    const float* wv   = (const float*)d_in[6];
    const float* bv   = (const float*)d_in[7];
    const float* wo   = (const float*)d_in[8];
    const float* bo   = (const float*)d_in[9];

    char* ws = (char*)d_ws;
    const size_t MB = 1u << 20;
    bf16* xb  = (bf16*)(ws + 0 * MB);
    bf16* wqb = (bf16*)(ws + 16 * MB);
    bf16* wkb = (bf16*)(ws + 18 * MB);
    bf16* wvb = (bf16*)(ws + 20 * MB);
    bf16* wob = (bf16*)(ws + 22 * MB);
    bf16* Qb  = (bf16*)(ws + 24 * MB);
    bf16* Kb  = (bf16*)(ws + 40 * MB);
    bf16* Vt  = (bf16*)(ws + 56 * MB);
    bf16* aob = (bf16*)(ws + 72 * MB);
    unsigned* mbits = (unsigned*)(ws + 88 * MB);
    int* rowfull    = (int*)(ws + 88 * MB + 512 * 1024);

    const int nx4 = (BATCH * SEQ * DMODEL) / 4;
    const int nw4 = (DMODEL * DMODEL) / 4;
    cvt_f32_bf16<<<nx4 / 256, 256, 0, stream>>>(x,  xb,  nx4);
    cvt_f32_bf16<<<nw4 / 256, 256, 0, stream>>>(wq, wqb, nw4);
    cvt_f32_bf16<<<nw4 / 256, 256, 0, stream>>>(wk, wkb, nw4);
    cvt_f32_bf16<<<nw4 / 256, 256, 0, stream>>>(wv, wvb, nw4);
    cvt_f32_bf16<<<nw4 / 256, 256, 0, stream>>>(wo, wob, nw4);

    pack_mask<<<(SEQ * SEQ) / 256, 256, 0, stream>>>(mask, mbits);
    mask_rowfull<<<(SEQ * 32) / 256, 256, 0, stream>>>(mbits, rowfull);

    dim3 gg(DMODEL / 128, (BATCH * SEQ) / 128);   // (8, 64)
    gemm_bf16<0><<<gg, 256, 0, stream>>>(xb, wqb, bq, (void*)Qb);
    gemm_bf16<1><<<gg, 256, 0, stream>>>(xb, wkb, bk, (void*)Kb);
    gemm_bf16<2><<<gg, 256, 0, stream>>>(xb, wvb, bv, (void*)Vt);

    dim3 ga(SEQ / 128, BATCH * NH);               // (16, 64)
    flash_attn<<<ga, 256, 0, stream>>>(Qb, Kb, Vt, mbits, rowfull, aob);

    gemm_bf16<3><<<gg, 256, 0, stream>>>(aob, wob, bo, d_out);
}
